// DeepFM_66468913873627
// MI455X (gfx1250) — compile-verified, matrix operand order checked by reading, not run on hardware
//
#include <hip/hip_runtime.h>
#include <hip/hip_bf16.h>

typedef __bf16 bf16;
typedef __attribute__((ext_vector_type(16))) __bf16 v16bf;
typedef __attribute__((ext_vector_type(8)))  __bf16 v8bf;
typedef __attribute__((ext_vector_type(4)))  __bf16 v4bf;
typedef __attribute__((ext_vector_type(8)))  float  v8f;

#define DIM    1024
#define BROWS  16384
#define NLAYER 4
#define LDSPAD 40   // row stride in bf16 (80B, 16B-aligned, breaks bank conflicts)

__device__ __forceinline__ v16bf cat8(v8bf lo, v8bf hi) {
  return __builtin_shufflevector(lo, hi, 0,1,2,3,4,5,6,7,8,9,10,11,12,13,14,15);
}

// ---------------------------------------------------------------------------
// prep: rowsum(x), interactions = relu(x * rowsum) -> bf16, x -> bf16
// ---------------------------------------------------------------------------
__global__ __launch_bounds__(256) void prep_kernel(const float* __restrict__ x,
                                                   bf16* __restrict__ xb,
                                                   bf16* __restrict__ inter) {
  const int row = blockIdx.x;
  const int t   = threadIdx.x;
  const float4 v = reinterpret_cast<const float4*>(x + row * DIM)[t];
  float s = v.x + v.y + v.z + v.w;

  __shared__ float red[256];
  red[t] = s;
  __syncthreads();
  #pragma unroll
  for (int off = 128; off > 0; off >>= 1) {
    if (t < off) red[t] += red[t + off];
    __syncthreads();
  }
  const float rs = red[0];

  const float xs[4] = {v.x, v.y, v.z, v.w};
  v4bf xo, io;
  #pragma unroll
  for (int i = 0; i < 4; ++i) {
    xo[i] = (bf16)xs[i];
    float iv = xs[i] * rs;
    io[i] = (bf16)(iv > 0.f ? iv : 0.f);
  }
  const int base4 = row * (DIM / 4) + t;
  reinterpret_cast<v4bf*>(xb)[base4]    = xo;
  reinterpret_cast<v4bf*>(inter)[base4] = io;
}

// ---------------------------------------------------------------------------
// f32 -> bf16 weight conversion
// ---------------------------------------------------------------------------
__global__ __launch_bounds__(256) void cvt_bf16_kernel(const float* __restrict__ src,
                                                       bf16* __restrict__ dst, int n4) {
  int i = blockIdx.x * 256 + threadIdx.x;
  if (i >= n4) return;
  float4 v = reinterpret_cast<const float4*>(src)[i];
  v4bf o;
  o[0] = (bf16)v.x; o[1] = (bf16)v.y; o[2] = (bf16)v.z; o[3] = (bf16)v.w;
  reinterpret_cast<v4bf*>(dst)[i] = o;
}

// ---------------------------------------------------------------------------
// C = epilogue(A @ W^T + bias)
// Block tile 128(M) x 256(N), BK=32. 8 wave32 waves in 2(M) x 4(N) grid,
// each wave owns 64x64 = 4x4 fragments of v_wmma_f32_16x16x32_bf16.
// Staging: GLOBAL_LOAD_ASYNC_TO_LDS_B128 (ASYNCcnt), double-buffered LDS.
//   mode 0: relu -> bf16        mode 1: (relu+inter)*0.5 -> bf16
//   mode 2: plain -> f32
// ---------------------------------------------------------------------------
__global__ __launch_bounds__(256) void gemm_kernel(
    const bf16*  __restrict__ A,
    const bf16*  __restrict__ W,
    const float* __restrict__ bias,
    bf16*        __restrict__ outb,
    float*       __restrict__ outf,
    const bf16*  __restrict__ inter,
    int mode) {
  __shared__ bf16 ldsA[2][128 * LDSPAD];
  __shared__ bf16 ldsB[2][256 * LDSPAD];

  const int t    = threadIdx.x;
  const int lane = t & 31;          // wave32
  const int wave = t >> 5;          // 0..7
  const int wm   = wave >> 2;       // 0..1 -> 64-row strip
  const int wn   = wave & 3;        // 0..3 -> 64-col strip
  const int m0   = blockIdx.y * 128;
  const int n0   = blockIdx.x * 256;

  // async staging: A tile 128x32 = 512 chunks of 8 bf16 (2/thread),
  //                B tile 256x32 = 1024 chunks (4/thread)
  auto issue = [&](int kk, int buf) {
    const int kb = kk * 32;
    #pragma unroll
    for (int j = 0; j < 2; ++j) {
      const int c = t + j * 256, row = c >> 2, kc = (c & 3) * 8;
      unsigned loff = (unsigned)(uintptr_t)&ldsA[buf][row * LDSPAD + kc];
      const bf16* g = A + (m0 + row) * DIM + kb + kc;
      asm volatile("global_load_async_to_lds_b128 %0, %1, off"
                   :: "v"(loff), "v"(g) : "memory");
    }
    #pragma unroll
    for (int j = 0; j < 4; ++j) {
      const int c = t + j * 256, row = c >> 2, kc = (c & 3) * 8;
      unsigned loff = (unsigned)(uintptr_t)&ldsB[buf][row * LDSPAD + kc];
      const bf16* g = W + (n0 + row) * DIM + kb + kc;
      asm volatile("global_load_async_to_lds_b128 %0, %1, off"
                   :: "v"(loff), "v"(g) : "memory");
    }
  };

  v8f acc[4][4] = {};

  issue(0, 0);

  // ISA 16-bit A/B fragment layout: lanes 0-15 hold K {0..7, 16..23},
  // lanes 16-31 hold K {8..15, 24..31}; row/col = lane & 15.
  const int ks   = (lane < 16) ? 0 : 8;
  const int lrow = lane & 15;

  for (int kk = 0; kk < DIM / 32; ++kk) {
    asm volatile("s_wait_asynccnt 0x0" ::: "memory");
    __syncthreads();
    const int cur = kk & 1;
    if (kk + 1 < DIM / 32) issue(kk + 1, cur ^ 1);

    v16bf af[4], bfr[4];
    #pragma unroll
    for (int im = 0; im < 4; ++im) {
      const bf16* p = &ldsA[cur][(wm * 64 + im * 16 + lrow) * LDSPAD + ks];
      af[im] = cat8(*reinterpret_cast<const v8bf*>(p),
                    *reinterpret_cast<const v8bf*>(p + 16));
    }
    #pragma unroll
    for (int jn = 0; jn < 4; ++jn) {
      const bf16* p = &ldsB[cur][(wn * 64 + jn * 16 + lrow) * LDSPAD + ks];
      bfr[jn] = cat8(*reinterpret_cast<const v8bf*>(p),
                     *reinterpret_cast<const v8bf*>(p + 16));
    }
    #pragma unroll
    for (int im = 0; im < 4; ++im)
      #pragma unroll
      for (int jn = 0; jn < 4; ++jn)
        acc[im][jn] = __builtin_amdgcn_wmma_f32_16x16x32_bf16(
            false, af[im], false, bfr[jn], (short)0, acc[im][jn], false, false);
  }

  // epilogue (C/D layout: lanes 0-15 -> M = vgpr, lanes 16-31 -> M = vgpr+8)
  const int mhi = (lane >> 4) << 3;
  #pragma unroll
  for (int im = 0; im < 4; ++im) {
    #pragma unroll
    for (int jn = 0; jn < 4; ++jn) {
      const int   gcol = n0 + wn * 64 + jn * 16 + lrow;
      const float bj   = bias[gcol];
      #pragma unroll
      for (int v = 0; v < 8; ++v) {
        const int grow = m0 + wm * 64 + im * 16 + mhi + v;
        const int idx  = grow * DIM + gcol;
        float z = acc[im][jn][v] + bj;
        if (mode == 0) {
          z = z > 0.f ? z : 0.f;
          outb[idx] = (bf16)z;
        } else if (mode == 1) {
          z = z > 0.f ? z : 0.f;
          z = (z + (float)inter[idx]) * 0.5f;
          outb[idx] = (bf16)z;
        } else {
          outf[idx] = z;
        }
      }
    }
  }
}

// ---------------------------------------------------------------------------
extern "C" void kernel_launch(void* const* d_in, const int* in_sizes, int n_in,
                              void* d_out, int out_size, void* d_ws, size_t ws_size,
                              hipStream_t stream) {
  (void)in_sizes; (void)n_in; (void)out_size; (void)ws_size;
  const float* x  = (const float*)d_in[0];
  const float* Ws = (const float*)d_in[1];
  const float* bs = (const float*)d_in[2];
  const float* Wo = (const float*)d_in[3];
  const float* bo = (const float*)d_in[4];
  float* out = (float*)d_out;

  char* ws = (char*)d_ws;
  const size_t BD2 = (size_t)BROWS * DIM * sizeof(bf16);   // 32 MB each
  bf16* xb    = (bf16*)ws;  ws += BD2;
  bf16* inter = (bf16*)ws;  ws += BD2;
  bf16* ha    = (bf16*)ws;  ws += BD2;
  bf16* hb    = (bf16*)ws;  ws += BD2;
  bf16* Wsb   = (bf16*)ws;  ws += (size_t)NLAYER * DIM * DIM * sizeof(bf16);
  bf16* Wob   = (bf16*)ws;

  prep_kernel<<<BROWS, 256, 0, stream>>>(x, xb, inter);

  int n4 = NLAYER * DIM * DIM / 4;
  cvt_bf16_kernel<<<(n4 + 255) / 256, 256, 0, stream>>>(Ws, Wsb, n4);
  n4 = DIM * DIM / 4;
  cvt_bf16_kernel<<<(n4 + 255) / 256, 256, 0, stream>>>(Wo, Wob, n4);

  dim3 grid(DIM / 256, BROWS / 128);  // (4, 128)
  gemm_kernel<<<grid, 256, 0, stream>>>(xb, Wsb,               bs,           ha, nullptr, nullptr, 0);
  gemm_kernel<<<grid, 256, 0, stream>>>(ha, Wsb + 1 * DIM*DIM, bs + 1 * DIM, hb, nullptr, nullptr, 0);
  gemm_kernel<<<grid, 256, 0, stream>>>(hb, Wsb + 2 * DIM*DIM, bs + 2 * DIM, ha, nullptr, nullptr, 0);
  gemm_kernel<<<grid, 256, 0, stream>>>(ha, Wsb + 3 * DIM*DIM, bs + 3 * DIM, hb, nullptr, inter,   1);
  gemm_kernel<<<grid, 256, 0, stream>>>(hb, Wob,               bo,           nullptr, out, nullptr, 2);
}
